// RNN_39376260170302
// MI455X (gfx1250) — matrix operand order repaired and checked by reading, not compiled, not run on hardware
//
#include <hip/hip_runtime.h>
#include <hip/hip_bf16.h>
#include <stdint.h>

#define VOCAB 32000
#define HID   1024
#define BATCH 256
#define SEQ   512
#define G3    (3 * HID)
#define NBLK_SCAN 64   // persistent WGs: conservative, guaranteed co-resident

// ---------------------------------------------------------------------------
// Types for WMMA fragments (CDNA5 wave32: v_wmma_f32_16x16x32_bf16)
// ---------------------------------------------------------------------------
typedef __attribute__((ext_vector_type(16))) __bf16        v16bf;
typedef __attribute__((ext_vector_type(8)))  float         v8f;
typedef __attribute__((ext_vector_type(4)))  unsigned int  u32x4;
typedef __attribute__((ext_vector_type(4)))  int           v4i;

// address-space qualified pointers for gfx1250 async-to-LDS builtin
typedef __attribute__((address_space(1))) v4i* gbl_v4i_ptr;
typedef __attribute__((address_space(3))) v4i* lds_v4i_ptr;

union FragAB {
  v16bf bf;
  u32x4 q[2];   // two 16-byte chunks = 16 bf16 values per lane
};

#if defined(__AMDGCN__) && __has_builtin(__builtin_amdgcn_global_load_async_to_lds_b128)
#define HAVE_ASYNC_LDS 1
#else
#define HAVE_ASYNC_LDS 0
#endif

#if defined(__AMDGCN__) && __has_builtin(__builtin_amdgcn_s_wait_asynccnt)
#define WAIT_ASYNC(n) __builtin_amdgcn_s_wait_asynccnt(n)
#else
#define WAIT_ASYNC(n) asm volatile("s_wait_asynccnt %0" ::"i"(n) : "memory")
#endif

static __device__ __forceinline__ v8f vzero8() {
  v8f v;
#pragma unroll
  for (int i = 0; i < 8; ++i) v[i] = 0.0f;
  return v;
}

static __device__ __forceinline__ unsigned short f32_to_bf16_rne(float f) {
  unsigned u = __float_as_uint(f);
  unsigned r = 0x7FFFu + ((u >> 16) & 1u);   // round-to-nearest-even
  return (unsigned short)((u + r) >> 16);
}
static __device__ __forceinline__ float bf16_to_f32(unsigned short h) {
  return __uint_as_float(((unsigned)h) << 16);
}

// ---------------------------------------------------------------------------
// Fragment loaders (generic address space: global or LDS source).
// A (16x32 bf16, row-major, contiguous k order, not ISA table order):
//   lanes 0-15 : K=k0..k0+15 ; lanes 16-31: K=k0+16..k0+31
// B (32x16 bf16) from N-major weights W[N][K] (== torch layout, out = x@W^T):
//   lanes 0-15 : col=c0+lane, K=k0..k0+15 ; lanes 16-31: K=k0+16..k0+31
// ---------------------------------------------------------------------------
static __device__ __forceinline__ FragAB load_fragA(const unsigned short* base_row,
                                                    int k0, int half) {
  FragAB a;
  a.q[0] = *(const u32x4*)(base_row + k0 + half * 16);
  a.q[1] = *(const u32x4*)(base_row + k0 + half * 16 + 8);
  return a;
}
static __device__ __forceinline__ FragAB load_fragB_g(const unsigned short* W,
                                                      int nrow, int k0, int half) {
  const unsigned short* p = W + (size_t)nrow * HID + k0 + half * 16;
  FragAB b;
  b.q[0] = *(const u32x4*)(p);
  b.q[1] = *(const u32x4*)(p + 8);
  return b;
}
// B fragment out of an LDS tile laid out [64 rows][32 k] bf16
static __device__ __forceinline__ FragAB load_fragB_lds(const unsigned short* tile16,
                                                        int nloc, int half) {
  const unsigned short* p = tile16 + nloc * 32 + half * 16;
  FragAB b;
  b.q[0] = *(const u32x4*)(p);
  b.q[1] = *(const u32x4*)(p + 8);
  return b;
}

// ---------------------------------------------------------------------------
// Prep kernels
// ---------------------------------------------------------------------------
__global__ void cvt_f32_to_bf16(const float* __restrict__ src,
                                unsigned short* __restrict__ dst, int n) {
  int i = blockIdx.x * blockDim.x + threadIdx.x;
  int stride = gridDim.x * blockDim.x;
  for (; i < n; i += stride) dst[i] = f32_to_bf16_rne(src[i]);
}

__global__ void init_state(float* __restrict__ hf, unsigned short* __restrict__ hb,
                           unsigned* __restrict__ ctr, int nh, int nc) {
  int i = blockIdx.x * blockDim.x + threadIdx.x;
  int stride = gridDim.x * blockDim.x;
  for (int j = i; j < nh; j += stride) { hf[j] = 0.0f; hb[j] = 0; }
  for (int j = i; j < nc; j += stride) { ctr[j] = 0; }
}

// ---------------------------------------------------------------------------
// Kernel 1: fused embedding gather + input projection GEMM.
// xg[s*B+b][g] = sum_k emb[inputs[b][s]][k] * W_ih[g][k] + b_ih[g]
// M = S*B = 131072, N = 3072, K = 1024, bf16 in / f32 acc / bf16 out.
// WG = 256 thr / 8 waves; tile = 128 rows x 64 cols; wave = 16-row stripe x 4
// column WMMA tiles. The 64x32 B tile is staged in LDS (double-buffered) via
// gfx1250 global_load_async_to_lds_b128 when available (ASYNCcnt-tracked);
// A rows gathered straight from global.
// ---------------------------------------------------------------------------
__global__ __launch_bounds__(256)
void emb_xgates_wmma(const int* __restrict__ inputs,            // [B][S]
                     const unsigned short* __restrict__ embb,   // [VOCAB][H] bf16
                     const unsigned short* __restrict__ Wib,    // [3H][H] bf16
                     const float* __restrict__ b_ih,            // [3H]
                     unsigned short* __restrict__ xg)           // [S*B][3H] bf16
{
  __shared__ v4i btile[2][256];   // 2 x (64 rows x 32 k) bf16 = 2 x 4KB

  const int tid  = threadIdx.x;
  const int wave = tid >> 5;
  const int lane = tid & 31;
  const int half = lane >> 4;
  const int l16  = lane & 15;

  const int c0 = blockIdx.x * 64;                 // column tile base in [0,3H)
  const int r0 = blockIdx.y * 128 + wave * 16;    // row tile base in [0,S*B)

  // staging slot for this thread: row = tid>>2 (0..63), 16B chunk = tid&3
  const unsigned short* bsrc_row = Wib + (size_t)(c0 + (tid >> 2)) * HID + (tid & 3) * 8;

  // this lane's A row: row r0+l16 -> (s,b) -> embedding row
  const int row   = r0 + l16;
  const int s     = row / BATCH;
  const int b     = row % BATCH;
  const int token = inputs[b * SEQ + s];
  const unsigned short* arow = embb + (size_t)token * HID;

  v8f acc[4] = { vzero8(), vzero8(), vzero8(), vzero8() };

  // ---- stage k0 = 0 into buffer 0
#if HAVE_ASYNC_LDS
  __builtin_amdgcn_global_load_async_to_lds_b128(
      (gbl_v4i_ptr)(uintptr_t)(bsrc_row),
      (lds_v4i_ptr)&btile[0][tid], 0, 0);
#else
  btile[0][tid] = *(const v4i*)(bsrc_row);
#endif

  for (int k0 = 0; k0 < HID; k0 += 32) {
    const int  cur  = (k0 >> 5) & 1;
    const bool more = (k0 + 32) < HID;
    if (more) {                                    // prefetch next B tile
#if HAVE_ASYNC_LDS
      __builtin_amdgcn_global_load_async_to_lds_b128(
          (gbl_v4i_ptr)(uintptr_t)(bsrc_row + k0 + 32),
          (lds_v4i_ptr)&btile[cur ^ 1][tid], 0, 0);
#else
      btile[cur ^ 1][tid] = *(const v4i*)(bsrc_row + k0 + 32);
#endif
    }
#if HAVE_ASYNC_LDS
    if (more) WAIT_ASYNC(1);
    else      WAIT_ASYNC(0);
#endif
    __syncthreads();                               // btile[cur] ready WG-wide

    if (k0 + 64 < HID) __builtin_prefetch(arow + k0 + 64, 0, 1);  // global_prefetch_b8
    FragAB A = load_fragA(arow, k0, half);
    const unsigned short* tile16 = (const unsigned short*)&btile[cur][0];
#pragma unroll
    for (int t = 0; t < 4; ++t) {
      FragAB B = load_fragB_lds(tile16, t * 16 + l16, half);
      acc[t] = __builtin_amdgcn_wmma_f32_16x16x32_bf16(
          false, A.bf, false, B.bf, (short)0, acc[t], false, false);
    }
    __syncthreads();                               // done reading before overwrite
  }

  // C/D layout: col = c0 + t*16 + l16 ; rows = r0 + half*8 + e
#pragma unroll
  for (int t = 0; t < 4; ++t) {
    const int col  = c0 + t * 16 + l16;
    const float bias = b_ih[col];
#pragma unroll
    for (int e = 0; e < 8; ++e) {
      const int rr = r0 + half * 8 + e;
      xg[(size_t)rr * G3 + col] = f32_to_bf16_rne(acc[t][e] + bias);
    }
  }
}

// ---------------------------------------------------------------------------
// Persistent GRU scan: ONE kernel runs all 512 steps; steps separated by a
// device-wide barrier (monotonic per-step counters -> no reset/ABA hazards).
// 64 WGs x 256 threads (co-resident on any plausible WGP count -> no
// deadlock). WG tile = 64 rows x 64 cols; wave w: rows (w>>1)*16, cols
// (w&1)*32 (2 subtiles) x 3 gates = 6 accumulators; A fragment reused 6x.
// W_hh (6MB bf16) stays L2-resident across all steps.
// ---------------------------------------------------------------------------
__device__ __forceinline__ void grid_barrier(unsigned* ctr_s, unsigned nblk) {
  __threadfence();                    // release h stores to device scope
  __syncthreads();
  if (threadIdx.x == 0) {
    __hip_atomic_fetch_add(ctr_s, 1u, __ATOMIC_ACQ_REL, __HIP_MEMORY_SCOPE_AGENT);
    while (__hip_atomic_load(ctr_s, __ATOMIC_ACQUIRE, __HIP_MEMORY_SCOPE_AGENT) < nblk)
      __builtin_amdgcn_s_sleep(2);
  }
  __syncthreads();
  __threadfence();                    // acquire: invalidate stale L0 lines
}

__global__ __launch_bounds__(256)
void gru_scan_persistent(float* __restrict__ hf0, unsigned short* __restrict__ hb0,
                         float* __restrict__ hf1, unsigned short* __restrict__ hb1,
                         const unsigned short* __restrict__ Whh,   // [3H][H] bf16
                         const float* __restrict__ b_hh,           // [3H]
                         const unsigned short* __restrict__ xg,    // [S*B][3H] bf16
                         unsigned* __restrict__ ctr)               // [SEQ] zeroed
{
  const int wave = threadIdx.x >> 5;
  const int lane = threadIdx.x & 31;
  const int half = lane >> 4;
  const int l16  = lane & 15;

  const int cw  = blockIdx.x & 15;    // 16 column tiles of 64
  const int rwb = blockIdx.x >> 4;    // 4 row tiles of 64
  const int mi  = wave >> 1;          // 0..3 : 16-row stripe
  const int ni2 = wave & 1;           // 0..1 : 32-col half

  const int r0    = rwb * 64 + mi * 16;
  const int cbase = cw * 64 + ni2 * 32;     // subtiles at cbase, cbase+16

  // per-lane invariant addresses
  const size_t arow_off = (size_t)(r0 + l16) * HID;
  float bh[3][2];
#pragma unroll
  for (int g = 0; g < 3; ++g)
#pragma unroll
    for (int c = 0; c < 2; ++c)
      bh[g][c] = b_hh[g * HID + cbase + c * 16 + l16];

  for (int s = 0; s < SEQ; ++s) {
    const float*          hpf = (s & 1) ? hf1 : hf0;
    const unsigned short* hpb = (s & 1) ? hb1 : hb0;
    float*                hnf = (s & 1) ? hf0 : hf1;
    unsigned short*       hnb = (s & 1) ? hb0 : hb1;
    const unsigned short* xg_s = xg + (size_t)s * BATCH * G3;

    v8f acc[3][2];
#pragma unroll
    for (int g = 0; g < 3; ++g) { acc[g][0] = vzero8(); acc[g][1] = vzero8(); }

    const unsigned short* arow = hpb + arow_off;
    for (int k0 = 0; k0 < HID; k0 += 32) {
      FragAB A = load_fragA(arow, k0, half);
#pragma unroll
      for (int c = 0; c < 2; ++c) {
#pragma unroll
        for (int g = 0; g < 3; ++g) {
          FragAB B = load_fragB_g(Whh, g * HID + cbase + c * 16 + l16, k0, half);
          acc[g][c] = __builtin_amdgcn_wmma_f32_16x16x32_bf16(
              false, A.bf, false, B.bf, (short)0, acc[g][c], false, false);
        }
      }
    }

#pragma unroll
    for (int c = 0; c < 2; ++c) {
      const int col = cbase + c * 16 + l16;
#pragma unroll
      for (int e = 0; e < 8; ++e) {
        const int    rr    = r0 + half * 8 + e;
        const size_t xbase = (size_t)rr * G3;
        const float xr = bf16_to_f32(xg_s[xbase + col]);
        const float xz = bf16_to_f32(xg_s[xbase + HID + col]);
        const float xn = bf16_to_f32(xg_s[xbase + 2 * HID + col]);
        const float hp = hpf[(size_t)rr * HID + col];

        const float r  = 1.0f / (1.0f + __expf(-(xr + acc[0][c][e] + bh[0][c])));
        const float z  = 1.0f / (1.0f + __expf(-(xz + acc[1][c][e] + bh[1][c])));
        const float n  = tanhf(xn + r * (acc[2][c][e] + bh[2][c]));
        const float hv = (1.0f - z) * n + z * hp;

        hnf[(size_t)rr * HID + col] = hv;
        hnb[(size_t)rr * HID + col] = f32_to_bf16_rne(hv);
      }
    }

    if (s + 1 < SEQ) grid_barrier(&ctr[s], NBLK_SCAN);
  }
}

// ---------------------------------------------------------------------------
// Final dense head: out[b][j] = h[b] . W_dense[j] + b_dense[j]
// ---------------------------------------------------------------------------
__global__ __launch_bounds__(256)
void dense_out(const float* __restrict__ h, const float* __restrict__ Wd,
               const float* __restrict__ bd, float* __restrict__ out) {
  __shared__ float red0[256];
  __shared__ float red1[256];
  const int b = blockIdx.x, t = threadIdx.x;
  float s0 = 0.0f, s1 = 0.0f;
  for (int k = t; k < HID; k += 256) {
    const float hv = h[(size_t)b * HID + k];
    s0 += hv * Wd[k];
    s1 += hv * Wd[HID + k];
  }
  red0[t] = s0; red1[t] = s1;
  __syncthreads();
  for (int off = 128; off > 0; off >>= 1) {
    if (t < off) { red0[t] += red0[t + off]; red1[t] += red1[t + off]; }
    __syncthreads();
  }
  if (t == 0) {
    out[b * 2 + 0] = red0[0] + bd[0];
    out[b * 2 + 1] = red1[0] + bd[1];
  }
}

// ---------------------------------------------------------------------------
// Host-side launch. Workspace (~886 MB):
// emb bf16 | W_ih bf16 | W_hh bf16 | xg bf16 | h f32 x2 | h bf16 x2 | ctr[SEQ]
// ---------------------------------------------------------------------------
extern "C" void kernel_launch(void* const* d_in, const int* in_sizes, int n_in,
                              void* d_out, int out_size, void* d_ws, size_t ws_size,
                              hipStream_t stream) {
  const int*   inputs = (const int*)  d_in[0];
  const float* emb    = (const float*)d_in[1];
  const float* W_ih   = (const float*)d_in[2];
  const float* W_hh   = (const float*)d_in[3];
  const float* b_ih   = (const float*)d_in[4];
  const float* b_hh   = (const float*)d_in[5];
  const float* W_d    = (const float*)d_in[6];
  const float* b_d    = (const float*)d_in[7];
  float* out = (float*)d_out;

  char* p = (char*)d_ws;
  auto carve = [&](size_t bytes) -> char* {
    char* r = p;
    p += (bytes + 255) & ~(size_t)255;
    return r;
  };
  unsigned short* embb = (unsigned short*)carve((size_t)VOCAB * HID * 2);
  unsigned short* Wib  = (unsigned short*)carve((size_t)G3 * HID * 2);
  unsigned short* Whb  = (unsigned short*)carve((size_t)G3 * HID * 2);
  unsigned short* xg   = (unsigned short*)carve((size_t)SEQ * BATCH * G3 * 2);
  float*          hf0  = (float*)carve((size_t)BATCH * HID * 4);
  float*          hf1  = (float*)carve((size_t)BATCH * HID * 4);
  unsigned short* hb0  = (unsigned short*)carve((size_t)BATCH * HID * 2);
  unsigned short* hb1  = (unsigned short*)carve((size_t)BATCH * HID * 2);
  unsigned*       ctr  = (unsigned*)carve((size_t)SEQ * 4);

  cvt_f32_to_bf16<<<2048, 256, 0, stream>>>(emb,  embb, VOCAB * HID);
  cvt_f32_to_bf16<<<1024, 256, 0, stream>>>(W_ih, Wib,  G3 * HID);
  cvt_f32_to_bf16<<<1024, 256, 0, stream>>>(W_hh, Whb,  G3 * HID);
  init_state<<<512, 256, 0, stream>>>(hf0, hb0, ctr, BATCH * HID, SEQ);

  dim3 g1(G3 / 64, (SEQ * BATCH) / 128);                    // 48 x 1024 WGs
  emb_xgates_wmma<<<g1, 256, 0, stream>>>(inputs, embb, Wib, b_ih, xg);

  gru_scan_persistent<<<NBLK_SCAN, 256, 0, stream>>>(hf0, hb0, hf1, hb1,
                                                     Whb, b_hh, xg, ctr);

  // s = 511 (odd) wrote hf0 -> final hidden state lives in buffer 0
  dense_out<<<BATCH, 256, 0, stream>>>(hf0, W_d, b_d, out);
}
